// EnhancedAIDetector_72507637891539
// MI455X (gfx1250) — matrix operand order repaired and checked by reading, not compiled, not run on hardware
//
#include <hip/hip_runtime.h>
#include <hip/hip_bf16.h>

// ---------------------------------------------------------------------------
// CDNA5 (gfx1250) EnhancedAIDetector.
// - All matmul tensors stored f16 in HBM (halves traffic; chip is HBM-bound).
// - Convs + GEMMs -> V_WMMA_F32_16X16X32_F16, 64x64x32 tiles, 8 wave32 waves.
// - GEMM A-tiles staged via Tensor Data Mover (tensor_load_to_lds +
//   s_wait_tensorcnt); B-tiles via 16B vector loads (K-contiguous layouts).
// ---------------------------------------------------------------------------

typedef __attribute__((ext_vector_type(16))) _Float16     v16h;
typedef __attribute__((ext_vector_type(8)))  _Float16     v8h;
typedef __attribute__((ext_vector_type(8)))  float        v8f;
typedef __attribute__((ext_vector_type(4)))  unsigned int u32x4;
typedef __attribute__((ext_vector_type(8)))  int          i32x8;
typedef __attribute__((ext_vector_type(4)))  int          i32x4;

#define TILE_M 64
#define TILE_N 64
#define TILE_K 32
#define LDA_S  40   // LDS row stride (halfs): 32 data + 8 pad (=4 DWORDs, TDM pad)
#define LDB_S  40
#define PI_F   3.14159265358979f

__device__ __forceinline__ v16h cat8(v8h lo, v8h hi) {
  return __builtin_shufflevector(lo, hi, 0, 1, 2, 3, 4, 5, 6, 7,
                                         8, 9, 10, 11, 12, 13, 14, 15);
}

// --------------------------- wave-level WMMA core --------------------------
// As[m][k] (64x32, stride LDA_S), Bs[n][k] (64x32, stride LDB_S), both f16.
// A frag: lanes 0-15 M=lane K=0..15, lanes 16-31 K=16..31 (K-contiguous);
// B frag: lanes 0-15 N=lane K=0..15, lanes 16-31 K=16..31 (K-contiguous).
__device__ __forceinline__ void wave_mma(const _Float16* __restrict__ As,
                                         const _Float16* __restrict__ Bs,
                                         int m0, int n0, v8f& acc0, v8f& acc1) {
  const int lane = threadIdx.x & 31;
  const int hf   = lane >> 4;
  const int l16  = lane & 15;
  const int m  = m0 + l16;
  const int kb = hf * 16;
  const v16h a = cat8(*(const v8h*)&As[m * LDA_S + kb],
                      *(const v8h*)&As[m * LDA_S + kb + 8]);
  const int kc = hf * 16;
  const int n  = n0 + l16;
  const v16h b0 = cat8(*(const v8h*)&Bs[n * LDB_S + kc],
                       *(const v8h*)&Bs[n * LDB_S + kc + 8]);
  const v16h b1 = cat8(*(const v8h*)&Bs[(n + 16) * LDB_S + kc],
                       *(const v8h*)&Bs[(n + 16) * LDB_S + kc + 8]);
  acc0 = __builtin_amdgcn_wmma_f32_16x16x32_f16(false, a, false, b0, (short)0,
                                                acc0, false, false);
  acc1 = __builtin_amdgcn_wmma_f32_16x16x32_f16(false, a, false, b1, (short)0,
                                                acc1, false, false);
}

// C frag: lane<16 -> N=lane rows m0..m0+7; lane>=16 -> N=lane-16 rows m0+8..15.
__device__ __forceinline__ void store_frag(_Float16* __restrict__ C,
                                           long long sc_r, long long sc_c,
                                           long long row_base, long long col_base,
                                           long long M, long long N,
                                           int m0, int n0,
                                           const float* __restrict__ bias,
                                           int relu, int acc, float alpha,
                                           const v8f& c) {
  const int lane = threadIdx.x & 31;
  const int hf   = lane >> 4;
  const int l16  = lane & 15;
  const long long gcol = col_base + n0 + l16;
  if (gcol >= N) return;
  const float bv = bias ? bias[gcol] : 0.0f;
#pragma unroll
  for (int v = 0; v < 8; ++v) {
    const long long grow = row_base + m0 + hf * 8 + v;
    if (grow < M) {
      float val = alpha * c[v] + bv;
      _Float16* p = C + grow * sc_r + gcol * sc_c;
      if (acc) val += (float)*p;
      if (relu) val = val > 0.0f ? val : 0.0f;
      *p = (_Float16)val;
    }
  }
}

// ----------------------------- batched GEMM --------------------------------
// C[g](MxN) = alpha * A[g](MxK, row-major, lda) * B[g](KxN) [+C][bias][relu].
// B is given K-contiguous: Bt[g][n*ldb + k] = B[k][n]. K must be a mult of 32.
__global__ __launch_bounds__(256) void k_gemm(
    const _Float16* __restrict__ A, long long a_batch, long long lda,
    const _Float16* __restrict__ Bt, long long b_batch, long long ldb,
    const float* __restrict__ bias,
    _Float16* __restrict__ C, long long c_batch, long long sc_r, long long sc_c,
    int M, int N, int K, int relu, int acc, float alpha) {
  __shared__ _Float16 As[TILE_M * LDA_S];
  __shared__ _Float16 Bs[TILE_N * LDB_S];
  const int tid = threadIdx.x;
  const _Float16* Ag = A + (long long)blockIdx.z * a_batch;
  const _Float16* Bg = Bt + (long long)blockIdx.z * b_batch;
  _Float16*       Cg = C + (long long)blockIdx.z * c_batch;
  const long long row_base = (long long)blockIdx.y * TILE_M;
  const long long col_base = (long long)blockIdx.x * TILE_N;

  const int wave = tid >> 5;
  const int wm0  = (wave & 3) * 16;
  const int wn0  = (wave >> 2) * 32;
  v8f acc0 = {}, acc1 = {};

  const int bn = tid & 63;            // B tile: n row per thread
  const int bk = (tid >> 6) * 8;      // 8 contiguous k per thread
  const long long gbn = col_base + bn;
  const int Mrem = (int)(M - row_base);

  for (int k0 = 0; k0 < K; k0 += TILE_K) {
#if __has_builtin(__builtin_amdgcn_tensor_load_to_lds)
    // --- A tile via Tensor Data Mover (one issue per block, wave 0) ---
    if (tid < 32) {
      const unsigned long long ga =
          (unsigned long long)(size_t)(const void*)(Ag + row_base * lda + k0);
      const unsigned lds_off = (unsigned)(size_t)(const void*)As;  // LDS byte off
      u32x4 g0;
      g0[0] = 1u;                                   // count=1, user descriptor
      g0[1] = lds_off;                              // lds_addr
      g0[2] = (unsigned)(ga & 0xffffffffu);         // global_addr[31:0]
      g0[3] = (unsigned)((ga >> 32) & 0x1ffffffu) | (2u << 30);  // addr[56:32]|type=2
      const int td0 = K - k0;                       // tensor_dim0 (elements left)
      const int td1 = Mrem;                         // tensor_dim1 (rows left)
      i32x8 g1;
      // data_size=2B, pad_enable, pad_interval=16 DW (32 halfs), pad=4 DW (8 halfs)
      g1[0] = (1 << 16) | (1 << 20) | (3 << 22) | (3 << 25);
      g1[1] = (td0 & 0xffff) << 16;
      g1[2] = ((td0 >> 16) & 0xffff) | ((td1 & 0xffff) << 16);
      g1[3] = ((td1 >> 16) & 0xffff) | (TILE_K << 16);  // tile_dim0 = 32
      g1[4] = TILE_M;                                   // tile_dim1 = 64
      g1[5] = (int)lda;                                 // tensor_dim0_stride
      g1[6] = 0;
      g1[7] = 0;
      i32x4 z4 = {0, 0, 0, 0};
#if __clang_major__ >= 23
      i32x8 z8 = {0, 0, 0, 0, 0, 0, 0, 0};
      __builtin_amdgcn_tensor_load_to_lds(g0, g1, z4, z4, z8, 0);
#else
      __builtin_amdgcn_tensor_load_to_lds(g0, g1, z4, z4, 0);
#endif
    }
#else
    // --- manual A tile fallback: one 16B load + one 16B LDS store/thread ---
    {
      const int ar = tid >> 2;
      const int ac = (tid & 3) * 8;
      v8h av = {};
      if (ar < Mrem) av = *(const v8h*)&Ag[(row_base + ar) * lda + k0 + ac];
      *(v8h*)&As[ar * LDA_S + ac] = av;
    }
#endif
    // --- B tile: one 16B load + one 16B LDS store per thread ---
    {
      v8h w8 = {};
      if (gbn < N) w8 = *(const v8h*)&Bg[gbn * ldb + k0 + bk];
      *(v8h*)&Bs[bn * LDB_S + bk] = w8;
    }
    if (k0 + TILE_K < K && gbn < N)
      __builtin_prefetch(&Bg[gbn * ldb + k0 + TILE_K + bk], 0, 0);
#if __has_builtin(__builtin_amdgcn_tensor_load_to_lds)
    if (tid < 32) __builtin_amdgcn_s_wait_tensorcnt((short)0);
#endif
    __syncthreads();
    wave_mma(As, Bs, wm0, wn0, acc0, acc1);
    __syncthreads();
  }
  store_frag(Cg, sc_r, sc_c, row_base, col_base, M, N, wm0, wn0,
             bias, relu, acc, alpha, acc0);
  store_frag(Cg, sc_r, sc_c, row_base, col_base, M, N, wm0, wn0 + 16,
             bias, relu, acc, alpha, acc1);
}

// --------------------------- implicit-GEMM conv ----------------------------
// f16 NCHW in, f16 weights pre-padded to [Cout][K32] (K32 = K rounded to 32).
// bcast3: input is one [Bn,H,W] plane replicated across input channels.
__global__ __launch_bounds__(256) void k_conv(
    const _Float16* __restrict__ in, int Bn, int Cin, int H, int W,
    const _Float16* __restrict__ w, const float* __restrict__ bias,
    int K, int K32, int kh, int kw, int stride, int pad,
    _Float16* __restrict__ out, int Cout, int Ho, int Wo, int bcast3) {
  __shared__ _Float16 As[TILE_M * LDA_S];
  __shared__ _Float16 Bs[TILE_N * LDB_S];
  const int tid  = threadIdx.x;
  const int HoWo = Ho * Wo;
  const long long M = (long long)Bn * HoWo;
  const long long row_base = (long long)blockIdx.y * TILE_M;
  const long long col_base = (long long)blockIdx.x * TILE_N;

  const int wave = tid >> 5;
  const int wm0  = (wave & 3) * 16;
  const int wn0  = (wave >> 2) * 32;
  v8f acc0 = {}, acc1 = {};

  // A staging: thread owns one output pixel row, 8 k's  (decode row once)
  const int ar = tid >> 2;
  const int ac = (tid & 3) * 8;
  const long long gm = row_base + ar;
  int b = 0, oy = 0, ox = 0;
  if (gm < M) {
    b = (int)(gm / HoWo);
    const int p = (int)(gm % HoWo);
    oy = p / Wo;
    ox = p % Wo;
  }
  // B staging: thread owns one Cout row, 8 contiguous (padded) k's
  const int bn = tid & 63;
  const int bk = (tid >> 6) * 8;
  const long long gbn = col_base + bn;

  for (int k0 = 0; k0 < K32; k0 += TILE_K) {
    {
      v8h av = {};
      if (gm < M) {
#pragma unroll
        for (int i = 0; i < 8; ++i) {
          const int gk = k0 + ac + i;
          if (gk < K) {
            const int cin = gk / (kh * kw);
            const int rr  = (gk / kw) % kh;
            const int ss  = gk % kw;
            const int iy = oy * stride - pad + rr;
            const int ix = ox * stride - pad + ss;
            if (iy >= 0 && iy < H && ix >= 0 && ix < W) {
              const long long plane = bcast3 ? (long long)b
                                             : ((long long)b * Cin + cin);
              av[i] = in[plane * ((long long)H * W) + (long long)iy * W + ix];
            }
          }
        }
      }
      *(v8h*)&As[ar * LDA_S + ac] = av;
    }
    {
      v8h w8 = {};
      if (gbn < Cout) w8 = *(const v8h*)&w[gbn * (long long)K32 + k0 + bk];
      *(v8h*)&Bs[bn * LDB_S + bk] = w8;
    }
    __syncthreads();
    wave_mma(As, Bs, wm0, wn0, acc0, acc1);
    __syncthreads();
  }

  // epilogue: scatter to f16 NCHW with bias + relu
  const int lane = tid & 31;
  const int hf   = lane >> 4;
  const int l16  = lane & 15;
#pragma unroll
  for (int t = 0; t < 2; ++t) {
    const v8f& c = t ? acc1 : acc0;
    const long long gcol = col_base + wn0 + 16 * t + l16;
    if (gcol >= Cout) continue;
    const float bv = bias[gcol];
#pragma unroll
    for (int v = 0; v < 8; ++v) {
      const long long grow = row_base + wm0 + hf * 8 + v;
      if (grow < M) {
        const int bb = (int)(grow / HoWo);
        const int pp = (int)(grow % HoWo);
        float val = c[v] + bv;
        val = val > 0.0f ? val : 0.0f;
        out[((long long)bb * Cout + gcol) * HoWo + pp] = (_Float16)val;
      }
    }
  }
}

// ------------------------------ small kernels ------------------------------
__global__ void k_zero(float* p, long long n) {
  long long i = (long long)blockIdx.x * blockDim.x + threadIdx.x;
  if (i < n) p[i] = 0.0f;
}

__global__ void k_cvt(const float* __restrict__ s, _Float16* __restrict__ d,
                      long long n) {
  long long i = (long long)blockIdx.x * blockDim.x + threadIdx.x;
  if (i < n) d[i] = (_Float16)s[i];
}

// conv weights: f32 [Cout][K] -> f16 [Cout][K32], zero padded
__global__ void k_cvt_pad(const float* __restrict__ s, int Cout, int K, int K32,
                          _Float16* __restrict__ d) {
  long long i = (long long)blockIdx.x * blockDim.x + threadIdx.x;
  if (i >= (long long)Cout * K32) return;
  const int n = (int)(i / K32), k = (int)(i % K32);
  d[i] = (_Float16)(k < K ? s[(long long)n * K + k] : 0.0f);
}

// FC weights: f32 [K][N] -> f16 [N][K] (K-contiguous for GEMM B side)
__global__ void k_cvt_T(const float* __restrict__ s, int K, int N,
                        _Float16* __restrict__ d) {
  long long i = (long long)blockIdx.x * blockDim.x + threadIdx.x;
  if (i >= (long long)K * N) return;
  const int k = (int)(i / N), n = (int)(i % N);
  d[(long long)n * K + k] = (_Float16)s[i];
}

// DFT tables (f16): C = cos(2 pi j k/224), Smat = -sin(...)  (W = C + i*Smat)
__global__ void k_build_dft(_Float16* dftC, _Float16* dftS) {
  int i = blockIdx.x * blockDim.x + threadIdx.x;
  if (i >= 224 * 224) return;
  const int j = i / 224, k = i % 224;
  const int m = (j * k) % 224;
  const float ang = 2.0f * PI_F * (float)m / 224.0f;
  dftC[i] = (_Float16)__cosf(ang);
  dftS[i] = (_Float16)(-__sinf(ang));
}

__device__ __forceinline__ float dct_coef(int i, int k) {
  float c = __cosf(PI_F * (2.0f * k + 1.0f) * i / 16.0f) * 0.5f;  // sqrt(2/8)
  if (i == 0) c *= 0.70710678f;
  return c;
}

// grayscale -> 8x8 block DCT-II(ortho) -> mean over 28x28 blocks (f32 acc)
__global__ void k_gray_dct(const float* __restrict__ x, float* __restrict__ dct_img) {
  const int blk = blockIdx.x;  // b*784 + by*28 + bx
  const int b = blk / 784;
  const int r = blk % 784;
  const int by = r / 28, bx = r % 28;
  const int i = threadIdx.x >> 3, j = threadIdx.x & 7;
  __shared__ float sb[8][8];
  __shared__ float st[8][8];
  const int y = by * 8 + i, xx = bx * 8 + j;
  const long long base = (long long)b * 3 * 50176;
  const float g = 0.299f * x[base + (long long)y * 224 + xx]
                + 0.587f * x[base + 50176 + (long long)y * 224 + xx]
                + 0.114f * x[base + 100352 + (long long)y * 224 + xx];
  sb[i][j] = g * 255.0f;
  __syncthreads();
  float t = 0.0f;
#pragma unroll
  for (int k = 0; k < 8; ++k) t += dct_coef(i, k) * sb[k][j];
  st[i][j] = t;
  __syncthreads();
  float o = 0.0f;
#pragma unroll
  for (int k = 0; k < 8; ++k) o += st[i][k] * dct_coef(j, k);
  atomicAdd(&dct_img[b * 64 + i * 8 + j], o * (1.0f / 784.0f));
}

// bilinear resize [B,8,8] f32 -> [B,224,224] f16 (half-pixel, edge clamp)
__global__ void k_resize(const float* __restrict__ dct_img,
                         _Float16* __restrict__ plane) {
  long long idx = (long long)blockIdx.x * blockDim.x + threadIdx.x;
  if (idx >= (long long)64 * 50176) return;
  const int b = (int)(idx / 50176);
  const int p = (int)(idx % 50176);
  const int oy = p / 224, ox = p % 224;
  float fy = (oy + 0.5f) * (8.0f / 224.0f) - 0.5f;
  float fx = (ox + 0.5f) * (8.0f / 224.0f) - 0.5f;
  fy = fminf(fmaxf(fy, 0.0f), 7.0f);
  fx = fminf(fmaxf(fx, 0.0f), 7.0f);
  const int y0 = (int)fy, x0 = (int)fx;
  const int y1 = min(y0 + 1, 7), x1 = min(x0 + 1, 7);
  const float wy = fy - y0, wx = fx - x0;
  const float* d = dct_img + b * 64;
  plane[idx] = (_Float16)((1.0f - wy) * ((1.0f - wx) * d[y0 * 8 + x0] + wx * d[y0 * 8 + x1])
                        + wy * ((1.0f - wx) * d[y1 * 8 + x0] + wx * d[y1 * 8 + x1]));
}

// global average pool f16 [64,C,HW] -> f16 out[b*out_stride + c]
__global__ void k_gap(const _Float16* __restrict__ in, int C, int HW,
                      _Float16* __restrict__ out, int out_stride) {
  int idx = blockIdx.x * blockDim.x + threadIdx.x;
  if (idx >= 64 * C) return;
  const int b = idx / C, c = idx % C;
  const _Float16* p = in + ((long long)b * C + c) * HW;
  float s = 0.0f;
  for (int i = 0; i < HW; ++i) s += (float)p[i];
  out[b * out_stride + c] = (_Float16)(s / (float)HW);
}

// 56x56 -> 7x7 mean pool, f16 [64,64,56,56] -> f16 cols of [64, out_stride]
__global__ void k_pool7(const _Float16* __restrict__ in, _Float16* __restrict__ out,
                        int out_stride, int col_off) {
  int idx = blockIdx.x * blockDim.x + threadIdx.x;
  if (idx >= 64 * 3136) return;
  const int b = idx / 3136;
  const int r = idx % 3136;  // c*49 + oy*7 + ox (matches _pool7 flatten)
  const int c = r / 49;
  const int q = r % 49;
  const int oy = q / 7, ox = q % 7;
  const _Float16* p = in + ((long long)b * 64 + c) * 3136;
  float s = 0.0f;
  for (int dy = 0; dy < 8; ++dy)
    for (int dx = 0; dx < 8; ++dx)
      s += (float)p[(oy * 8 + dy) * 56 + ox * 8 + dx];
  out[b * out_stride + col_off + r] = (_Float16)(s * (1.0f / 64.0f));
}

// classifier f16[64,512] @ f32[512,2] + bias, temperature; d_out f32
__global__ void k_cls(const _Float16* __restrict__ fused, const float* __restrict__ Wc,
                      const float* __restrict__ bc, const float* __restrict__ temp,
                      float* __restrict__ out) {
  const int t = threadIdx.x;
  if (t >= 128) return;
  const int b = t >> 1, j = t & 1;
  float s = bc[j];
  const _Float16* f = fused + b * 512;
  for (int k = 0; k < 512; ++k) s += (float)f[k] * Wc[k * 2 + j];
  out[b * 2 + j] = s;
  out[128 + b * 2 + j] = s / temp[0];
}

// ------------------------------- launcher ----------------------------------
extern "C" void kernel_launch(void* const* d_in, const int* in_sizes, int n_in,
                              void* d_out, int out_size, void* d_ws, size_t ws_size,
                              hipStream_t stream) {
  (void)in_sizes; (void)n_in; (void)out_size; (void)ws_size;
  const float* x      = (const float*)d_in[0];
  const float* w_stem = (const float*)d_in[1];  const float* b_stem = (const float*)d_in[2];
  const float* w1     = (const float*)d_in[3];  const float* b1     = (const float*)d_in[4];
  const float* w2     = (const float*)d_in[5];  const float* b2     = (const float*)d_in[6];
  const float* w3     = (const float*)d_in[7];  const float* b3     = (const float*)d_in[8];
  const float* wd1    = (const float*)d_in[9];  const float* bd1    = (const float*)d_in[10];
  const float* wd2    = (const float*)d_in[11]; const float* bd2    = (const float*)d_in[12];
  const float* wf1    = (const float*)d_in[13]; const float* bf1    = (const float*)d_in[14];
  const float* wf2    = (const float*)d_in[15]; const float* bf2    = (const float*)d_in[16];
  const float* W_freq = (const float*)d_in[17]; const float* b_freq = (const float*)d_in[18];
  const float* W_fu1  = (const float*)d_in[19]; const float* b_fu1  = (const float*)d_in[20];
  const float* W_fu2  = (const float*)d_in[21]; const float* b_fu2  = (const float*)d_in[22];
  const float* W_cls  = (const float*)d_in[23]; const float* b_cls  = (const float*)d_in[24];
  const float* temp   = (const float*)d_in[25];
  float* out = (float*)d_out;

  // f16 arena (16B-aligned chunks); bigA/bigB reused across branches
  _Float16* ws = (_Float16*)d_ws;
  size_t off = 0;
  auto ah = [&](size_t n) { _Float16* p = ws + off; off += (n + 7) & ~(size_t)7; return p; };
  _Float16* x_h     = ah((size_t)64 * 3 * 50176);
  _Float16* dftC    = ah(50176);
  _Float16* dftS    = ah(50176);
  _Float16* TrT     = ah((size_t)192 * 50176);   // (X@C)^T per image
  _Float16* TiT     = ah((size_t)192 * 50176);   // (X@Smat)^T per image
  _Float16* fft_in  = ah((size_t)64 * 6 * 50176);
  _Float16* dct_pl  = ah((size_t)64 * 50176);
  _Float16* h2      = ah((size_t)64 * 256 * 28 * 28);
  _Float16* h3      = ah((size_t)64 * 512 * 14 * 14);
  _Float16* freqcat = ah((size_t)64 * 6272);
  _Float16* comb    = ah((size_t)64 * 1024);
  _Float16* fu1o    = ah((size_t)64 * 1024);
  _Float16* fused   = ah((size_t)64 * 512);
  _Float16* bigA    = ah((size_t)64 * 64 * 112 * 112);   // stem / hd1 / hf1
  _Float16* bigB    = ah((size_t)64 * 128 * 56 * 56);    // h1 / hd2 / hf2
  _Float16* ws_h    = ah((size_t)64 * 160);              // stem w, K=147->160
  _Float16* w1_h    = ah((size_t)128 * 576);
  _Float16* w2_h    = ah((size_t)256 * 1152);
  _Float16* w3_h    = ah((size_t)512 * 2304);
  _Float16* wd1_h   = ah((size_t)32 * 32);               // K=27->32
  _Float16* wd2_h   = ah((size_t)64 * 288);
  _Float16* wf1_h   = ah((size_t)32 * 64);               // K=54->64
  _Float16* wf2_h   = ah((size_t)64 * 288);
  _Float16* Wfreq_T = ah((size_t)512 * 6272);
  _Float16* Wfu1_T  = ah((size_t)1024 * 1024);
  _Float16* Wfu2_T  = ah((size_t)512 * 1024);
  float*    dct_img = (float*)ah(2 * 64 * 64);           // f32 [64,8,8]

  auto cdiv = [](long long a, long long b) { return (unsigned)((a + b - 1) / b); };
  auto cvtpad = [&](const float* s, int Cout, int K, int K32, _Float16* d) {
    k_cvt_pad<<<cdiv((long long)Cout * K32, 256), 256, 0, stream>>>(s, Cout, K, K32, d);
  };
  auto conv = [&](const _Float16* in, int Cin, int H, int W, const _Float16* w,
                  int K, int K32, const float* b, int kh, int kw, int s, int pad,
                  _Float16* o, int Cout, int Ho, int Wo, int bc3) {
    const long long M = (long long)64 * Ho * Wo;
    dim3 g(cdiv(Cout, 64), cdiv(M, 64), 1);
    k_conv<<<g, 256, 0, stream>>>(in, 64, Cin, H, W, w, b, K, K32, kh, kw, s, pad,
                                  o, Cout, Ho, Wo, bc3);
  };

  // ---- precision conversion passes (one-time per call, tiny vs main work) --
  k_cvt<<<cdiv((long long)64 * 3 * 50176, 256), 256, 0, stream>>>(x, x_h, (long long)64 * 3 * 50176);
  cvtpad(w_stem, 64, 147, 160, ws_h);
  cvtpad(w1, 128, 576, 576, w1_h);
  cvtpad(w2, 256, 1152, 1152, w2_h);
  cvtpad(w3, 512, 2304, 2304, w3_h);
  cvtpad(wd1, 32, 27, 32, wd1_h);
  cvtpad(wd2, 64, 288, 288, wd2_h);
  cvtpad(wf1, 32, 54, 64, wf1_h);
  cvtpad(wf2, 64, 288, 288, wf2_h);
  k_cvt_T<<<cdiv((long long)6272 * 512, 256), 256, 0, stream>>>(W_freq, 6272, 512, Wfreq_T);
  k_cvt_T<<<cdiv((long long)1024 * 1024, 256), 256, 0, stream>>>(W_fu1, 1024, 1024, Wfu1_T);
  k_cvt_T<<<cdiv((long long)1024 * 512, 256), 256, 0, stream>>>(W_fu2, 1024, 512, Wfu2_T);
  k_build_dft<<<cdiv(50176, 256), 256, 0, stream>>>(dftC, dftS);
  k_zero<<<cdiv(64 * 64, 256), 256, 0, stream>>>(dct_img, 64 * 64);

  // ---- backbone: 224 ->112 ->56 ->28 ->14, GAP -> comb[:, 0:512] ----------
  conv(x_h,  3,   224, 224, ws_h, 147, 160, b_stem, 7, 7, 2, 2, bigA, 64,  112, 112, 0);
  conv(bigA, 64,  112, 112, w1_h, 576, 576, b1,     3, 3, 2, 0, bigB, 128, 56,  56,  0);
  conv(bigB, 128, 56,  56,  w2_h, 1152, 1152, b2,   3, 3, 2, 0, h2,   256, 28,  28,  0);
  conv(h2,   256, 28,  28,  w3_h, 2304, 2304, b3,   3, 3, 2, 0, h3,   512, 14,  14,  0);
  k_gap<<<cdiv(64 * 512, 256), 256, 0, stream>>>(h3, 512, 196, comb, 1024);

  // ---- DCT branch -> freqcat[:, 0:3136] -----------------------------------
  k_gray_dct<<<64 * 784, 64, 0, stream>>>(x, dct_img);
  k_resize<<<cdiv((long long)64 * 50176, 256), 256, 0, stream>>>(dct_img, dct_pl);
  conv(dct_pl, 3,  224, 224, wd1_h, 27, 32, bd1, 3, 3, 2, 0, bigA, 32, 112, 112, 1);
  conv(bigA,   32, 112, 112, wd2_h, 288, 288, bd2, 3, 3, 2, 0, bigB, 64, 56, 56, 0);
  k_pool7<<<cdiv(64 * 3136, 256), 256, 0, stream>>>(bigB, freqcat, 6272, 0);

  // ---- FFT branch: Z = W X W, 6 batched 224^3 WMMA GEMMs (g = b*3+c) ------
  {
    dim3 g(4, 4, 192);
    // stage 1: T = X @ (C + i*Smat); store transposed so stage 2 B is K-contig
    k_gemm<<<g, 256, 0, stream>>>(x_h, 50176, 224, dftC, 0, 224, nullptr,
                                  TrT, 50176, 1, 224, 224, 224, 224, 0, 0, 1.0f);
    k_gemm<<<g, 256, 0, stream>>>(x_h, 50176, 224, dftS, 0, 224, nullptr,
                                  TiT, 50176, 1, 224, 224, 224, 224, 0, 0, 1.0f);
    // stage 2: Zr = C@Tr - Smat@Ti ; Zi = C@Ti + Smat@Tr  (A = DFT matrix)
    k_gemm<<<g, 256, 0, stream>>>(dftC, 0, 224, TrT, 50176, 224, nullptr,
                                  fft_in, 100352, 224, 1, 224, 224, 224, 0, 0, 1.0f);
    k_gemm<<<g, 256, 0, stream>>>(dftS, 0, 224, TiT, 50176, 224, nullptr,
                                  fft_in, 100352, 224, 1, 224, 224, 224, 0, 1, -1.0f);
    k_gemm<<<g, 256, 0, stream>>>(dftC, 0, 224, TiT, 50176, 224, nullptr,
                                  fft_in + 50176, 100352, 224, 1, 224, 224, 224, 0, 0, 1.0f);
    k_gemm<<<g, 256, 0, stream>>>(dftS, 0, 224, TrT, 50176, 224, nullptr,
                                  fft_in + 50176, 100352, 224, 1, 224, 224, 224, 0, 1, 1.0f);
  }
  conv(fft_in, 6,  224, 224, wf1_h, 54, 64, bf1, 3, 3, 2, 0, bigA, 32, 112, 112, 0);
  conv(bigA,   32, 112, 112, wf2_h, 288, 288, bf2, 3, 3, 2, 0, bigB, 64, 56, 56, 0);
  k_pool7<<<cdiv(64 * 3136, 256), 256, 0, stream>>>(bigB, freqcat, 6272, 3136);

  // ---- fusion MLPs (WMMA GEMMs, K-contiguous transposed weights) ----------
  k_gemm<<<dim3(8, 1, 1), 256, 0, stream>>>(freqcat, 0, 6272, Wfreq_T, 0, 6272,
                                            b_freq, comb + 512, 0, 1024, 1,
                                            64, 512, 6272, 1, 0, 1.0f);
  k_gemm<<<dim3(16, 1, 1), 256, 0, stream>>>(comb, 0, 1024, Wfu1_T, 0, 1024,
                                             b_fu1, fu1o, 0, 1024, 1,
                                             64, 1024, 1024, 1, 0, 1.0f);
  k_gemm<<<dim3(8, 1, 1), 256, 0, stream>>>(fu1o, 0, 1024, Wfu2_T, 0, 1024,
                                            b_fu2, fused, 0, 512, 1,
                                            64, 512, 1024, 1, 0, 1.0f);
  k_cls<<<1, 128, 0, stream>>>(fused, W_cls, b_cls, temp, out);
}
